// MultiHeadAttention_89670327206616
// MI455X (gfx1250) — matrix operand order repaired and checked by reading, not compiled, not run on hardware
//
#include <hip/hip_runtime.h>
#include <hip/hip_bf16.h>
#include <math.h>

// MI455X / gfx1250, wave32. GEMMs run on V_WMMA_F32_16X16X32_BF16 (the high-rate
// matrix pipe) using split-precision bf16x3 emulation (hi/lo) for ~fp32 accuracy:
//   x ~= hi(x) + lo(x);  a*b ~= ah*bh + ah*bl + al*bh   (3 WMMAs per K=32 panel)
//
// Fragment layouts (wave32, from CDNA5 ISA 7.12.2):
//   A 16x32 bf16: lane M=lane&15; VGPR v holds K = 2v + (v>=4 ? 8 : 0) + 8*(lane>=16), pair (K,K+1)
//   B 32x16 bf16: lane N=lane&15; VGPR v holds K = 2v + (v>=4 ? 8 : 0) + 8*(lane>=16), pair (K,K+1)
//   C/D 16x16 f32: VGPR r -> M = r + 8*(lane>=16), N = lane&15

typedef __attribute__((ext_vector_type(2)))  float  v2f;
typedef __attribute__((ext_vector_type(8)))  float  v8f;
typedef __attribute__((ext_vector_type(16))) __bf16 v16bf;

union Frag { unsigned u[8]; v16bf v; };

#define WMMA_BF16(acc, a, b)                                                   \
  (acc) = __builtin_amdgcn_wmma_f32_16x16x32_bf16(false, (a), false, (b),      \
                                                  (short)0, (acc), false, false)

#define BATCH 8
#define SEQ   1024
#define DM    512
#define NH    8
#define DK    64

__device__ inline void split2(float x, __bf16& h, __bf16& l) {
  h = (__bf16)x;
  l = (__bf16)(x - (float)h);
}
__device__ inline unsigned pk(__bf16 a, __bf16 b) {
  union { __bf16 h[2]; unsigned u; } t;
  t.h[0] = a; t.h[1] = b;
  return t.u;
}

// ---------------------------------------------------------------------------
// GEMM: Y = X @ W^T + bias (X:[M,512] f32, W:[512,512] f32).
// Block tile 128(M) x 64(N), 8 waves, each wave 16x64 (4 C tiles), K panels of 32.
// mode 0: Yf = f32 [M,512]
// mode 1: Yh/Yl = bf16 hi/lo, head-split [B,H,S,DK]
// mode 2: Yh/Yl = bf16 hi/lo, head-split transposed [B,H,DK,S]   (for V)
// ---------------------------------------------------------------------------
__global__ __launch_bounds__(256) void mha_proj_kernel(
    const float* __restrict__ X, const float* __restrict__ W,
    const float* __restrict__ bias, float* __restrict__ Yf,
    __bf16* __restrict__ Yh, __bf16* __restrict__ Yl, int mode) {
  __shared__ __align__(16) __bf16 As_h[128][36];  // [m][k], stride 36 keeps uint pairs 4B-aligned
  __shared__ __align__(16) __bf16 As_l[128][36];
  __shared__ __align__(16) __bf16 Bs_h[64][36];   // [n][k] = W[n0+n][kc+k]
  __shared__ __align__(16) __bf16 Bs_l[64][36];

  const int m0 = blockIdx.x * 128;
  const int n0 = blockIdx.y * 64;
  const int tid = threadIdx.x;
  const int w = tid >> 5;
  const int lane = tid & 31;
  const int hi = lane >> 4;
  const int l15 = lane & 15;

  v8f acc[4] = {};

  for (int kc = 0; kc < DM; kc += 32) {
    // stage X panel 128x32 -> split hi/lo bf16 (1024 float4 chunks / 256 thr = 4 ea)
    for (int c = 0; c < 4; ++c) {
      const int chunk = tid + c * 256;
      const int row = chunk >> 3;
      const int k4 = (chunk & 7) * 4;
      float4 xv = *(const float4*)(X + (size_t)(m0 + row) * DM + kc + k4);
      __bf16 h0, l0, h1, l1, h2, l2, h3, l3;
      split2(xv.x, h0, l0); split2(xv.y, h1, l1);
      split2(xv.z, h2, l2); split2(xv.w, h3, l3);
      unsigned* ph = (unsigned*)&As_h[row][k4];
      unsigned* pl = (unsigned*)&As_l[row][k4];
      ph[0] = pk(h0, h1); ph[1] = pk(h2, h3);
      pl[0] = pk(l0, l1); pl[1] = pk(l2, l3);
    }
    // stage W panel 64x32 -> Bs[n][k] (coalesced row reads; no transpose needed)
    for (int c = 0; c < 2; ++c) {
      const int chunk = tid + c * 256;
      const int n = chunk >> 3;
      const int k4 = (chunk & 7) * 4;
      float4 wv = *(const float4*)(W + (size_t)(n0 + n) * DM + kc + k4);
      __bf16 h0, l0, h1, l1, h2, l2, h3, l3;
      split2(wv.x, h0, l0); split2(wv.y, h1, l1);
      split2(wv.z, h2, l2); split2(wv.w, h3, l3);
      unsigned* ph = (unsigned*)&Bs_h[n][k4];
      unsigned* pl = (unsigned*)&Bs_l[n][k4];
      ph[0] = pk(h0, h1); ph[1] = pk(h2, h3);
      pl[0] = pk(l0, l1); pl[1] = pk(l2, l3);
    }
    __syncthreads();

    Frag ah, al;
    const int arow = w * 16 + l15;
    for (int v = 0; v < 8; ++v) {
      const int kA = 2 * v + (v >= 4 ? 8 : 0) + 8 * hi;
      ah.u[v] = *(const unsigned*)&As_h[arow][kA];
      al.u[v] = *(const unsigned*)&As_l[arow][kA];
    }
    for (int j = 0; j < 4; ++j) {
      Frag bhf, blf;
      const int brow = j * 16 + l15;
      for (int v = 0; v < 8; ++v) {
        const int kB = 2 * v + (v >= 4 ? 8 : 0) + 8 * hi;
        bhf.u[v] = *(const unsigned*)&Bs_h[brow][kB];
        blf.u[v] = *(const unsigned*)&Bs_l[brow][kB];
      }
      WMMA_BF16(acc[j], ah.v, bhf.v);
      WMMA_BF16(acc[j], ah.v, blf.v);
      WMMA_BF16(acc[j], al.v, bhf.v);
    }
    __syncthreads();
  }

  for (int j = 0; j < 4; ++j) {
    for (int r = 0; r < 8; ++r) {
      const int row = m0 + w * 16 + r + hi * 8;
      const int col = n0 + j * 16 + l15;
      const float val = acc[j][r] + bias[col];
      if (mode == 0) {
        Yf[(size_t)row * DM + col] = val;
      } else {
        const int bb = row >> 10, srow = row & (SEQ - 1);
        const int hh = col >> 6, d = col & (DK - 1);
        __bf16 hv, lv;
        split2(val, hv, lv);
        const size_t idx = (mode == 1)
            ? ((((size_t)bb * NH + hh) * SEQ + srow) * DK + d)
            : ((((size_t)bb * NH + hh) * DK + d) * SEQ + srow);
        Yh[idx] = hv; Yl[idx] = lv;
      }
    }
  }
}

// ---------------------------------------------------------------------------
// Fused attention: one block per (b, h, 16-query tile).
// Phase A: scores = Q@K^T via bf16x3 WMMA -> 64KB LDS fp32 tile.
// Phase B: softmax -> cumsum -> distance decay -> softmax, in-place (fp32).
// Phase C: attn @ V (V pre-transposed [B,H,DK,S]) via bf16x3 WMMA.
// ---------------------------------------------------------------------------
__global__ __launch_bounds__(256) void mha_attn_kernel(
    const __bf16* __restrict__ Qh, const __bf16* __restrict__ Ql,
    const __bf16* __restrict__ Kh, const __bf16* __restrict__ Kl,
    const __bf16* __restrict__ Vth, const __bf16* __restrict__ Vtl,
    const float* __restrict__ gammas, const unsigned char* __restrict__ kpm,
    float* __restrict__ Cc) {
  __shared__ float sc[16][SEQ];  // 64 KB score/probability tile

  const int qt0 = blockIdx.x * 16;
  const int h = blockIdx.y;
  const int bb = blockIdx.z;
  const int bh = bb * NH + h;
  const int tid = threadIdx.x;
  const int w = tid >> 5;
  const int lane = tid & 31;
  const int hi = lane >> 4;
  const int l15 = lane & 15;

  // Q fragments for both K=32 panels (hi & lo) live in registers
  Frag aqh[2], aql[2];
  {
    const __bf16* qh_row = Qh + ((size_t)bh * SEQ + qt0 + l15) * DK;
    const __bf16* ql_row = Ql + ((size_t)bh * SEQ + qt0 + l15) * DK;
    for (int p = 0; p < 2; ++p)
      for (int v = 0; v < 8; ++v) {
        const int k = 32 * p + 2 * v + (v >= 4 ? 8 : 0) + 8 * hi;
        aqh[p].u[v] = *(const unsigned*)(qh_row + k);
        aql[p].u[v] = *(const unsigned*)(ql_row + k);
      }
  }

  // ---- Phase A: each wave owns 128 keys = 8 n-tiles ----
  for (int nt = 0; nt < 8; ++nt) {
    const int n0 = w * 128 + nt * 16;
    const __bf16* kh_row = Kh + ((size_t)bh * SEQ + n0 + l15) * DK;
    const __bf16* kl_row = Kl + ((size_t)bh * SEQ + n0 + l15) * DK;
    v8f acc = {};
    for (int p = 0; p < 2; ++p) {
      Frag bhf, blf;
      for (int v = 0; v < 8; ++v) {
        const int k = 32 * p + 2 * v + (v >= 4 ? 8 : 0) + 8 * hi;
        bhf.u[v] = *(const unsigned*)(kh_row + k);
        blf.u[v] = *(const unsigned*)(kl_row + k);
      }
      WMMA_BF16(acc, aqh[p].v, bhf.v);
      WMMA_BF16(acc, aqh[p].v, blf.v);
      WMMA_BF16(acc, aql[p].v, bhf.v);
    }
    for (int r = 0; r < 8; ++r) sc[r + 8 * hi][n0 + l15] = acc[r];
  }
  __syncthreads();

  // ---- Phase B: per-row decay pipeline (wave w owns rows w and w+8) ----
  const float gv = gammas[h];
  const float gamma = -((gv > 20.0f) ? gv : log1pf(__expf(gv)));  // -softplus
  const float scale = 0.125f;                                     // 1/sqrt(64)

  for (int rr = 0; rr < 2; ++rr) {
    const int i = w + rr * 8;
    const int qg = qt0 + i;
    const int j0 = lane * 32;

    unsigned vmask = 0;
    for (int u = 0; u < 32; ++u) {
      const int j = j0 + u;
      if ((j <= qg) && (kpm[bb * SEQ + j] == 0)) vmask |= (1u << u);
    }
    float mx = -3.0e38f;
    for (int u = 0; u < 32; ++u)
      if ((vmask >> u) & 1u) mx = fmaxf(mx, sc[i][j0 + u] * scale);
    for (int off = 16; off > 0; off >>= 1) mx = fmaxf(mx, __shfl_xor(mx, off, 32));
    float den = 0.0f;
    for (int u = 0; u < 32; ++u)
      if ((vmask >> u) & 1u) den += __expf(sc[i][j0 + u] * scale - mx);
    for (int off = 16; off > 0; off >>= 1) den += __shfl_xor(den, off, 32);
    const float invden = 1.0f / den;
    float ltot = 0.0f;
    for (int u = 0; u < 32; ++u)
      if ((vmask >> u) & 1u) ltot += __expf(sc[i][j0 + u] * scale - mx) * invden;
    float pre = ltot;
    for (int off = 1; off < 32; off <<= 1) {
      float t = __shfl_up(pre, off, 32);
      if (lane >= off) pre += t;
    }
    float run = pre - ltot;
    for (int u = 0; u < 32; ++u) {
      const int j = j0 + u;
      const float s = sc[i][j];
      float s2 = -3.0e38f;
      if ((vmask >> u) & 1u) {
        run += __expf(s * scale - mx) * invden;
        const float rem = 1.0f - run;
        const float dist = sqrtf(fmaxf(rem * fabsf((float)(qg - j)), 0.0f));
        float eff = __expf(dist * gamma);
        eff = fminf(fmaxf(eff, 1e-5f), 1e5f);
        s2 = s * scale * eff;
      }
      sc[i][j] = s2;
    }
    float mx2 = -3.0e38f;
    for (int u = 0; u < 32; ++u)
      if ((vmask >> u) & 1u) mx2 = fmaxf(mx2, sc[i][j0 + u]);
    for (int off = 16; off > 0; off >>= 1) mx2 = fmaxf(mx2, __shfl_xor(mx2, off, 32));
    float den2 = 0.0f;
    for (int u = 0; u < 32; ++u)
      if ((vmask >> u) & 1u) den2 += __expf(sc[i][j0 + u] - mx2);
    for (int off = 16; off > 0; off >>= 1) den2 += __shfl_xor(den2, off, 32);
    const float invden2 = 1.0f / den2;
    for (int u = 0; u < 32; ++u) {
      const int j = j0 + u;
      sc[i][j] = ((vmask >> u) & 1u) ? __expf(sc[i][j] - mx2) * invden2 : 0.0f;
    }
  }
  __syncthreads();

  // ---- Phase C: out = attn(16x1024) @ V(1024x64); waves 0..3, one 16-col d tile each
  if (w < 4) {
    const int d0 = w * 16;
    const __bf16* vth = Vth + ((size_t)bh * DK + d0 + l15) * SEQ;  // row = d, cols = keys
    const __bf16* vtl = Vtl + ((size_t)bh * DK + d0 + l15) * SEQ;
    v8f acc = {};
    for (int kp = 0; kp < 32; ++kp) {
      Frag ah, al, bhf, blf;
      for (int v = 0; v < 8; ++v) {
        const int kA = 32 * kp + 2 * v + (v >= 4 ? 8 : 0) + 8 * hi;
        v2f f = *(const v2f*)&sc[l15][kA];
        __bf16 h0, l0, h1, l1;
        split2(f.x, h0, l0); split2(f.y, h1, l1);
        ah.u[v] = pk(h0, h1); al.u[v] = pk(l0, l1);
        const int kB = 32 * kp + 2 * v + (v >= 4 ? 8 : 0) + 8 * hi;
        bhf.u[v] = *(const unsigned*)(vth + kB);
        blf.u[v] = *(const unsigned*)(vtl + kB);
      }
      WMMA_BF16(acc, ah.v, bhf.v);
      WMMA_BF16(acc, ah.v, blf.v);
      WMMA_BF16(acc, al.v, bhf.v);
    }
    for (int r = 0; r < 8; ++r) {
      const int qrow = qt0 + r + 8 * hi;
      const int col = h * DK + d0 + l15;  // concat layout [B,S,H*DK]
      Cc[((size_t)bb * SEQ + qrow) * DM + col] = acc[r];
    }
  }
}

// ---------------------------------------------------------------------------
extern "C" void kernel_launch(void* const* d_in, const int* in_sizes, int n_in,
                              void* d_out, int out_size, void* d_ws, size_t ws_size,
                              hipStream_t stream) {
  (void)in_sizes; (void)n_in; (void)out_size; (void)ws_size;
  const float* q   = (const float*)d_in[0];
  const float* k   = (const float*)d_in[1];
  const float* v   = (const float*)d_in[2];
  const float* Wq  = (const float*)d_in[3];
  const float* bq  = (const float*)d_in[4];
  const float* Wk  = (const float*)d_in[5];
  const float* bk  = (const float*)d_in[6];
  const float* Wv  = (const float*)d_in[7];
  const float* bv  = (const float*)d_in[8];
  const float* Wo  = (const float*)d_in[9];
  const float* bo  = (const float*)d_in[10];
  const float* gam = (const float*)d_in[11];
  const unsigned char* kpm = (const unsigned char*)d_in[13];  // [12]=causal mask (implicit)

  const size_t tsz = (size_t)BATCH * NH * SEQ * DK;  // 4M elements
  __bf16* Qh  = (__bf16*)d_ws;
  __bf16* Ql  = Qh + tsz;
  __bf16* Kh  = Ql + tsz;
  __bf16* Kl  = Kh + tsz;
  __bf16* Vth = Kl + tsz;
  __bf16* Vtl = Vth + tsz;
  float*  Cc  = (float*)(Vtl + tsz);  // [B,S,512] concat, f32

  const dim3 gblk(256);
  const dim3 gproj(BATCH * SEQ / 128, DM / 64, 1);
  hipLaunchKernelGGL(mha_proj_kernel, gproj, gblk, 0, stream, q, Wq, bq,
                     (float*)nullptr, Qh, Ql, 1);
  hipLaunchKernelGGL(mha_proj_kernel, gproj, gblk, 0, stream, k, Wk, bk,
                     (float*)nullptr, Kh, Kl, 1);
  hipLaunchKernelGGL(mha_proj_kernel, gproj, gblk, 0, stream, v, Wv, bv,
                     (float*)nullptr, Vth, Vtl, 2);

  const dim3 gattn(SEQ / 16, NH, BATCH);
  hipLaunchKernelGGL(mha_attn_kernel, gattn, gblk, 0, stream, Qh, Ql, Kh, Kl,
                     Vth, Vtl, gam, kpm, Cc);

  hipLaunchKernelGGL(mha_proj_kernel, gproj, gblk, 0, stream, Cc, Wo, bo,
                     (float*)d_out, (__bf16*)nullptr, (__bf16*)nullptr, 0);
}
